// GCTN_69595650064677
// MI455X (gfx1250) — matrix operand order repaired and checked by reading, not compiled, not run on hardware
//
#include <hip/hip_runtime.h>
#include <hip/hip_bf16.h>

typedef __attribute__((ext_vector_type(16))) _Float16 v16h;
typedef __attribute__((ext_vector_type(8)))  _Float16 v8h;
typedef __attribute__((ext_vector_type(8)))  float    v8f;

#define NTILES 52      // 832/16 output columns: Q(256)|K(256)|V(256)|skip(64)
#define NGROUP 13      // 4 N-tiles per wave
#define OUTW   832

// ---------------- CSR build ----------------

__global__ void k_zero_int(int* p, int n) {
    int i = blockIdx.x * 256 + threadIdx.x;
    if (i < n) p[i] = 0;
}

__global__ void k_count(const int* __restrict__ ei, int* __restrict__ cnt, int E) {
    int e = blockIdx.x * 256 + threadIdx.x;
    if (e < E) atomicAdd(&cnt[ei[E + e]], 1);
}

// single-block exclusive scan (N ~ 50k), 1024 threads
__global__ void k_scan_excl(const int* __restrict__ cnt, int* __restrict__ off, int n) {
    __shared__ int sh[1024];
    __shared__ int carry;
    int tid = threadIdx.x;
    if (tid == 0) carry = 0;
    __syncthreads();
    for (int base = 0; base < n; base += 1024) {
        int v = (base + tid < n) ? cnt[base + tid] : 0;
        sh[tid] = v;
        __syncthreads();
        for (int o = 1; o < 1024; o <<= 1) {
            int t = (tid >= o) ? sh[tid - o] : 0;
            __syncthreads();
            sh[tid] += t;
            __syncthreads();
        }
        if (base + tid < n) off[base + tid] = carry + sh[tid] - v;
        __syncthreads();
        if (tid == 0) carry += sh[1023];
        __syncthreads();
    }
    if (tid == 0) off[n] = carry;
}

__global__ void k_copy_int(const int* __restrict__ a, int* __restrict__ b, int n) {
    int i = blockIdx.x * 256 + threadIdx.x;
    if (i < n) b[i] = a[i];
}

__global__ void k_fill(const int* __restrict__ ei, const float* __restrict__ ea,
                       int* __restrict__ cur, int* __restrict__ key,
                       int* __restrict__ dst_s, int* __restrict__ src_s,
                       float* __restrict__ ea_s, int E) {
    int e = blockIdx.x * 256 + threadIdx.x;
    if (e >= E) return;
    int d = ei[E + e];
    int pos = atomicAdd(&cur[d], 1);
    key[pos] = e;
    dst_s[pos] = d;
    src_s[pos] = ei[e];
    ea_s[pos] = ea[e];
}

// deterministic order inside each segment: insertion-sort by original edge id
__global__ void k_sort_seg(const int* __restrict__ off, int* __restrict__ key,
                           int* __restrict__ src_s, float* __restrict__ ea_s, int N) {
    int n = blockIdx.x * 256 + threadIdx.x;
    if (n >= N) return;
    int beg = off[n], end = off[n + 1];
    for (int i = beg + 1; i < end; ++i) {
        int k = key[i]; int s = src_s[i]; float e = ea_s[i];
        int j = i - 1;
        while (j >= beg && key[j] > k) {
            key[j + 1] = key[j]; src_s[j + 1] = src_s[j]; ea_s[j + 1] = ea_s[j]; --j;
        }
        key[j + 1] = k; src_s[j + 1] = s; ea_s[j + 1] = e;
    }
}

// ---------------- dense phase ----------------

// pack Wq|Wk|Wv|Wskip (f32, row-major [64,cols]) into WMMA B-fragment layout (f16):
// Wfrag[((nt*2+s)*32+lane)*16 + j] holds W[K = (j&7) + 8*(lane>=16) + 16*(j>=8) + 32*s][nt*16 + (lane&15)]
__global__ void k_pack_w(const float* __restrict__ Wq, const float* __restrict__ Wk,
                         const float* __restrict__ Wv, const float* __restrict__ Ws,
                         _Float16* __restrict__ Wfrag) {
    int idx = blockIdx.x * 256 + threadIdx.x;
    if (idx >= NTILES * 2 * 32 * 16) return;
    int j    = idx & 15;
    int lane = (idx >> 4) & 31;
    int s    = (idx >> 9) & 1;
    int nt   = idx >> 10;
    int K    = (j & 7) + ((lane >> 4) << 3) + ((j >> 3) << 4) + (s << 5);
    int col  = nt * 16 + (lane & 15);
    float w;
    if      (col < 256) w = Wq[K * 256 + col];
    else if (col < 512) w = Wk[K * 256 + (col - 256)];
    else if (col < 768) w = Wv[K * 256 + (col - 512)];
    else                w = Ws[K * 64 + (col - 768)];
    Wfrag[idx] = (_Float16)w;
}

__global__ void k_pack_bias(const float* __restrict__ bq, const float* __restrict__ bk,
                            const float* __restrict__ bv, const float* __restrict__ bs,
                            float* __restrict__ ball) {
    int i = blockIdx.x * 256 + threadIdx.x;
    if (i >= OUTW) return;
    float b;
    if      (i < 256) b = bq[i];
    else if (i < 512) b = bk[i - 256];
    else if (i < 768) b = bv[i - 512];
    else              b = bs[i - 768];
    ball[i] = b;
}

__global__ void k_convert(const float* __restrict__ hin, _Float16* __restrict__ hf16,
                          int N, int N16) {
    int idx = blockIdx.x * 256 + threadIdx.x;
    if (idx >= N16 * 64) return;
    int row = idx >> 6;
    hf16[idx] = (row < N) ? (_Float16)hin[idx] : (_Float16)0.0f;
}

// fused Q|K|V|skip GEMM: [N16,64] x [64,832].
// One wave computes a 16-row strip x 4 consecutive 16-col tiles:
// A fragment loaded once (both K-steps), 8 v_wmma_f32_16x16x32_f16 per wave.
__global__ void k_gemm_qkvs(const _Float16* __restrict__ A, const _Float16* __restrict__ Wfrag,
                            const float* __restrict__ bias,
                            float* __restrict__ Q, float* __restrict__ Kb,
                            float* __restrict__ Vb, float* __restrict__ Sb, int mtiles) {
    int wave = blockIdx.x * (blockDim.x >> 5) + (threadIdx.x >> 5);
    int total = mtiles * NGROUP;
    if (wave >= total) return;                 // wave-uniform: EXEC all-1s inside wave
    int mt = wave % mtiles;
    int ng = wave / mtiles;                    // group of 4 N-tiles
    int lane = threadIdx.x & 31;
    int rlo = lane & 15;
    int hi  = lane >> 4;

    // A fragments for K-step 0 and 1 (16-bit A 16x32 layout)
    const _Float16* arow = A + (size_t)(mt * 16 + rlo) * 64;
    union { v16h v; v8h h[2]; } a0, a1;
    a0.h[0] = *(const v8h*)(arow + hi * 8);            // K =  8*hi .. +7
    a0.h[1] = *(const v8h*)(arow + hi * 8 + 16);       // K = 16+8*hi .. +7
    a1.h[0] = *(const v8h*)(arow + 32 + hi * 8);
    a1.h[1] = *(const v8h*)(arow + 32 + hi * 8 + 16);

    v8f c0 = {}, c1 = {}, c2 = {}, c3 = {};
    const _Float16* wbase = Wfrag + ((size_t)(ng * 4) * 2 * 32 + lane) * 16;
#pragma unroll
    for (int t = 0; t < 4; ++t) {
        v16h b0 = *(const v16h*)(wbase + (size_t)(t * 2 + 0) * 32 * 16);
        v16h b1 = *(const v16h*)(wbase + (size_t)(t * 2 + 1) * 32 * 16);
        v8f* cp = (t == 0) ? &c0 : (t == 1) ? &c1 : (t == 2) ? &c2 : &c3;
        *cp = __builtin_amdgcn_wmma_f32_16x16x32_f16(
                  false, a0.v, false, b0, (short)0, *cp, false, false);
        *cp = __builtin_amdgcn_wmma_f32_16x16x32_f16(
                  false, a1.v, false, b1, (short)0, *cp, false, false);
    }

    int mbase = mt * 16 + hi * 8;
#pragma unroll
    for (int t = 0; t < 4; ++t) {
        const v8f& c = (t == 0) ? c0 : (t == 1) ? c1 : (t == 2) ? c2 : c3;
        int col = (ng * 4 + t) * 16 + rlo;
        float bb = bias[col];
        float* dst; int ld, cc;
        if      (col < 256) { dst = Q;  cc = col;        ld = 256; }
        else if (col < 512) { dst = Kb; cc = col - 256;  ld = 256; }
        else if (col < 768) { dst = Vb; cc = col - 512;  ld = 256; }
        else                { dst = Sb; cc = col - 768;  ld = 64;  }
#pragma unroll
        for (int r = 0; r < 8; ++r)
            dst[(size_t)(mbase + r) * ld + cc] = c[r] + bb;
    }
}

// ---------------- edge phase ----------------

// one thread per (sorted-edge, head): alpha = q[dst]·(k[src] + ea*We)/8
__global__ void k_alpha(const float* __restrict__ Q, const float* __restrict__ Kb,
                        const int* __restrict__ dst_s, const int* __restrict__ src_s,
                        const float* __restrict__ ea_s, const float* __restrict__ We,
                        float* __restrict__ alpha, int E4) {
    int idx = blockIdx.x * 256 + threadIdx.x;
    if (idx >= E4) return;
    int p = idx >> 2, h = idx & 3;
    int d = dst_s[p], s = src_s[p];
    float ea = ea_s[p];
    const float4* q4 = (const float4*)(Q  + (size_t)d * 256 + h * 64);
    const float4* k4 = (const float4*)(Kb + (size_t)s * 256 + h * 64);
    const float4* w4 = (const float4*)(We + h * 64);
    float acc = 0.f;
#pragma unroll
    for (int i = 0; i < 16; ++i) {
        float4 q = q4[i], k = k4[i], w = w4[i];
        acc += q.x * (k.x + ea * w.x) + q.y * (k.y + ea * w.y)
             + q.z * (k.z + ea * w.z) + q.w * (k.w + ea * w.w);
    }
    alpha[idx] = acc * 0.125f;  // 1/sqrt(64)
}

// one wave per node: segment softmax + weighted message sum + head mean + skip (+res/relu)
__global__ void k_aggregate(const int* __restrict__ off, const int* __restrict__ src_s,
                            const float* __restrict__ ea_s, const float* __restrict__ alpha,
                            const float* __restrict__ Vb, const float* __restrict__ We,
                            const float* __restrict__ Sb, const float* __restrict__ resid,
                            float* __restrict__ out, int N, int mode) {
    int wid = blockIdx.x * (blockDim.x >> 5) + (threadIdx.x >> 5);
    if (wid >= N) return;
    int lane = threadIdx.x & 31;
    int beg = off[wid], end = off[wid + 1];
    float acc0 = 0.f, acc1 = 0.f;
#pragma unroll
    for (int h = 0; h < 4; ++h) {
        // pass 1: segment max (lanes parallel over edges)
        float m = -1e30f;
        for (int p = beg + lane; p < end; p += 32) m = fmaxf(m, alpha[p * 4 + h]);
#pragma unroll
        for (int o = 16; o > 0; o >>= 1) m = fmaxf(m, __shfl_xor(m, o, 32));
        // pass 2: lanes parallel over feature dim (d = lane, lane+32), serial edges
        float w0 = We[h * 64 + lane], w1 = We[h * 64 + lane + 32];
        float denom = 0.f, t0 = 0.f, t1 = 0.f;
        for (int p = beg; p < end; ++p) {
            if (p + 1 < end) {   // hide L2 gather latency behind the FMA chain
                const float* vnext = Vb + (size_t)src_s[p + 1] * 256 + h * 64;
                __builtin_prefetch(vnext + lane, 0, 0);     // global_prefetch_b8
            }
            float a = __expf(alpha[p * 4 + h] - m);
            denom += a;
            int s = src_s[p]; float ea = ea_s[p];
            const float* vrow = Vb + (size_t)s * 256 + h * 64;
            t0 += a * (vrow[lane]      + ea * w0);
            t1 += a * (vrow[lane + 32] + ea * w1);
        }
        float inv = 1.f / (denom + 1e-16f);
        acc0 += t0 * inv;
        acc1 += t1 * inv;
    }
    float v0 = acc0 * 0.25f + Sb[(size_t)wid * 64 + lane];
    float v1 = acc1 * 0.25f + Sb[(size_t)wid * 64 + lane + 32];
    if (mode == 0) {  // residual + relu (layers 0,1)
        v0 = fmaxf(v0 + resid[(size_t)wid * 64 + lane],      0.f);
        v1 = fmaxf(v1 + resid[(size_t)wid * 64 + lane + 32], 0.f);
    }
    out[(size_t)wid * 64 + lane]      = v0;
    out[(size_t)wid * 64 + lane + 32] = v1;
}

// ---------------- host side ----------------

static inline int cdiv(int a, int b) { return (a + b - 1) / b; }

extern "C" void kernel_launch(void* const* d_in, const int* in_sizes, int n_in,
                              void* d_out, int out_size, void* d_ws, size_t ws_size,
                              hipStream_t stream) {
    const int N   = in_sizes[0] / 64;
    const int E   = in_sizes[1] / 2;
    const int N16 = (N + 15) & ~15;
    const int mtiles = N16 / 16;

    const float* x  = (const float*)d_in[0];
    const int*   ei = (const int*)  d_in[1];
    const float* ea = (const float*)d_in[2];
    // params: per layer l (l=0..2): Wq,bq,Wk,bk,Wv,bv,We,Wskip,bskip at d_in[3+9l ..]

    // workspace carve-up
    char* w = (char*)d_ws;
    size_t off = 0;
    auto alloc = [&](size_t bytes) -> void* {
        void* p = w + off;
        off = (off + bytes + 255) & ~(size_t)255;
        return p;
    };
    _Float16* hf16  = (_Float16*)alloc((size_t)N16 * 64 * 2);
    _Float16* Wfrag = (_Float16*)alloc((size_t)NTILES * 2 * 32 * 16 * 2);
    float* ball  = (float*)alloc(OUTW * 4);
    float* Qb    = (float*)alloc((size_t)N16 * 256 * 4);
    float* Kb    = (float*)alloc((size_t)N16 * 256 * 4);
    float* Vb    = (float*)alloc((size_t)N16 * 256 * 4);
    float* Sb    = (float*)alloc((size_t)N16 * 64 * 4);
    float* alph  = (float*)alloc((size_t)E * 4 * 4);
    int*   offs  = (int*)  alloc((size_t)(N + 1) * 4);
    int*   cur   = (int*)  alloc((size_t)N * 4);
    int*   key   = (int*)  alloc((size_t)E * 4);
    int*   dst_s = (int*)  alloc((size_t)E * 4);
    int*   src_s = (int*)  alloc((size_t)E * 4);
    float* ea_s  = (float*)alloc((size_t)E * 4);
    float* hbuf0 = (float*)alloc((size_t)N * 64 * 4);
    float* hbuf1 = (float*)alloc((size_t)N * 64 * 4);
    (void)ws_size; (void)n_in;

    // ---- CSR build (edge_index invariant across layers) ----
    k_zero_int<<<cdiv(N, 256), 256, 0, stream>>>(cur, N);
    k_count<<<cdiv(E, 256), 256, 0, stream>>>(ei, cur, E);
    k_scan_excl<<<1, 1024, 0, stream>>>(cur, offs, N);
    k_copy_int<<<cdiv(N, 256), 256, 0, stream>>>(offs, cur, N);
    k_fill<<<cdiv(E, 256), 256, 0, stream>>>(ei, ea, cur, key, dst_s, src_s, ea_s, E);
    k_sort_seg<<<cdiv(N, 256), 256, 0, stream>>>(offs, key, src_s, ea_s, N);

    // ---- 3 layers ----
    const float* hin = x;
    for (int l = 0; l < 3; ++l) {
        const float* Wq = (const float*)d_in[3 + 9 * l + 0];
        const float* bq = (const float*)d_in[3 + 9 * l + 1];
        const float* Wk = (const float*)d_in[3 + 9 * l + 2];
        const float* bk = (const float*)d_in[3 + 9 * l + 3];
        const float* Wv = (const float*)d_in[3 + 9 * l + 4];
        const float* bv = (const float*)d_in[3 + 9 * l + 5];
        const float* We = (const float*)d_in[3 + 9 * l + 6];
        const float* Ws = (const float*)d_in[3 + 9 * l + 7];
        const float* bs = (const float*)d_in[3 + 9 * l + 8];

        k_pack_w<<<cdiv(NTILES * 2 * 32 * 16, 256), 256, 0, stream>>>(Wq, Wk, Wv, Ws, Wfrag);
        k_pack_bias<<<cdiv(OUTW, 256), 256, 0, stream>>>(bq, bk, bv, bs, ball);
        k_convert<<<cdiv(N16 * 64, 256), 256, 0, stream>>>(hin, hf16, N, N16);

        int totalWaves = mtiles * NGROUP;
        k_gemm_qkvs<<<cdiv(totalWaves, 8), 256, 0, stream>>>(hf16, Wfrag, ball,
                                                             Qb, Kb, Vb, Sb, mtiles);

        k_alpha<<<cdiv(E * 4, 256), 256, 0, stream>>>(Qb, Kb, dst_s, src_s, ea_s,
                                                      We, alph, E * 4);

        float* hout = (l == 0) ? hbuf0 : (l == 1) ? hbuf1 : (float*)d_out;
        int mode = (l < 2) ? 0 : 1;
        k_aggregate<<<cdiv(N, 8), 256, 0, stream>>>(offs, src_s, ea_s, alph, Vb, We,
                                                    Sb, hin, hout, N, mode);
        hin = hout;
    }
    (void)out_size;
}
